// EpisodicMemoryEfficient_89618787598704
// MI455X (gfx1250) — compile-verified, matrix operand order checked by reading, not run on hardware
//
#include <hip/hip_runtime.h>
#include <hip/hip_bf16.h>

typedef __attribute__((ext_vector_type(16))) __bf16 v16bf;
typedef __attribute__((ext_vector_type(8)))  float  v8f;

#define B_    4
#define S_    4096
#define D2_   512
#define MTOT  (B_ * S_)          /* 16384 rows */
#define BAND  64
#define SCALE 0.04419417382415922f  /* 1/sqrt(512) */

__device__ __forceinline__ unsigned short f2bf(float f) {
  unsigned int u = __float_as_uint(f);
  u += 0x7FFFu + ((u >> 16) & 1u);            // round-to-nearest-even
  return (unsigned short)(u >> 16);
}

// ---------------------------------------------------------------------------
// Kernel 1: convert Wq, Wk (f32 [512,512]) -> bf16
// ---------------------------------------------------------------------------
__global__ __launch_bounds__(256) void cvt_w_kernel(const float* __restrict__ wq,
                                                    const float* __restrict__ wk,
                                                    unsigned short* __restrict__ wqb,
                                                    unsigned short* __restrict__ wkb) {
  const int n4 = (D2_ * D2_) / 4;             // 65536 float4 per matrix
  int idx = blockIdx.x * blockDim.x + threadIdx.x;
  if (idx >= 2 * n4) return;
  const float4* src;
  unsigned short* dst;
  int i = idx;
  if (i < n4) { src = (const float4*)wq; dst = wqb; }
  else        { src = (const float4*)wk; dst = wkb; i -= n4; }
  float4 v = src[i];
  ushort4 o;
  o.x = f2bf(v.x); o.y = f2bf(v.y); o.z = f2bf(v.z); o.w = f2bf(v.w);
  *(ushort4*)(dst + (size_t)i * 4) = o;
}

// ---------------------------------------------------------------------------
// Kernel 2: x (f32 [MTOT,512]) -> bf16 row-major AND bf16 transposed [512,MTOT]
// LDS-tiled 32x32 transpose so both store streams are coalesced.
// ---------------------------------------------------------------------------
__global__ __launch_bounds__(256) void cvt_x_tr_kernel(const float* __restrict__ x,
                                                       unsigned short* __restrict__ xb,
                                                       unsigned short* __restrict__ xtb) {
  __shared__ unsigned short tile[32][33];
  const int m0 = blockIdx.x * 32;
  const int d0 = blockIdx.y * 32;
  const int tx = threadIdx.x, ty = threadIdx.y;
#pragma unroll
  for (int s = 0; s < 4; ++s) {
    int row = ty + s * 8;
    unsigned short bv = f2bf(x[(size_t)(m0 + row) * D2_ + d0 + tx]);
    xb[(size_t)(m0 + row) * D2_ + d0 + tx] = bv;
    tile[row][tx] = bv;
  }
  __syncthreads();
#pragma unroll
  for (int s = 0; s < 4; ++s) {
    int dr = ty + s * 8;
    xtb[(size_t)(d0 + dr) * MTOT + m0 + tx] = tile[tx][dr];
  }
}

// ---------------------------------------------------------------------------
// Kernel 3: q/k projection GEMM via bf16 WMMA.
//   grid.x = MTOT/16 (16-row tiles), grid.y = {0:q, 1:k}, 128 thr = 4 waves.
//   Each wave owns a 16x128 strip. W chunks (128 rows x 32 k = 8 KB) are
//   double-buffered in LDS via GLOBAL_LOAD_ASYNC_TO_LDS_B128 (ASYNCcnt),
//   B-fragments read from LDS, A-fragments software-pipelined from global.
//   Output strip bounced through LDS for 256B-contiguous stores.
// ---------------------------------------------------------------------------
__global__ __launch_bounds__(128) void gemm_qk_kernel(const unsigned short* __restrict__ xb,
                                                      const unsigned short* __restrict__ wqb,
                                                      const unsigned short* __restrict__ wkb,
                                                      unsigned short* __restrict__ qb,
                                                      unsigned short* __restrict__ kb) {
  __shared__ __align__(16) unsigned short wlds[4][2][4096];  // [wave][buf][128n x 32k] bf16
  __shared__ __align__(16) unsigned short outb[4][16][136];  // 16 x 128 strip + pad

  const unsigned short* W   = blockIdx.y ? wkb : wqb;
  unsigned short*       out = blockIdx.y ? kb  : qb;
  const int wave = threadIdx.x >> 5;
  const int lane = threadIdx.x & 31;
  const int half = lane >> 4;
  const int nn   = lane & 15;
  const size_t m0 = (size_t)blockIdx.x * 16;
  const int n0w = wave * 128;

  const unsigned long long wbase = (unsigned long long)W;
  // low 32 bits of a generic shared-pointer == wave-relative LDS byte offset
  const unsigned ldsbase = (unsigned)(uintptr_t)&wlds[wave][0][0];
  // flat chunk byte o = it*512 + lane*16 ; row n = o/64 ; in-row byte = o%64
  const unsigned gbase = (unsigned)(n0w * 1024) + ((unsigned)(lane >> 2) * 1024u)
                       + ((unsigned)(lane & 3) * 16u);

  auto fill = [&](int kc, int buf) {
    unsigned lofs = ldsbase + (unsigned)buf * 8192u + (unsigned)lane * 16u;
    unsigned gofs = gbase + (unsigned)kc * 64u;
#pragma unroll
    for (int it = 0; it < 16; ++it)
      asm volatile("global_load_async_to_lds_b128 %0, %1, %2"
                   :: "v"(lofs + (unsigned)it * 512u),
                      "v"(gofs + (unsigned)it * 8192u),
                      "s"(wbase)
                   : "memory");
  };

  v8f acc[8];
  {
    v8f z = {};
#pragma unroll
    for (int t = 0; t < 8; ++t) acc[t] = z;
  }

  fill(0, 0);
  v16bf a = *(const v16bf*)(xb + (m0 + nn) * D2_ + half * 16);

  for (int kc = 0; kc < 16; ++kc) {
    v16bf anext = a;
    if (kc < 15) {
      anext = *(const v16bf*)(xb + (m0 + nn) * D2_ + (kc + 1) * 32 + half * 16);
      fill(kc + 1, (kc + 1) & 1);
      // 16 newest fills may be pending; in-order completion => chunk kc resident
      asm volatile("s_wait_asynccnt 0x10" ::: "memory");
    } else {
      asm volatile("s_wait_asynccnt 0x0" ::: "memory");
    }
    const unsigned short* wb = &wlds[wave][kc & 1][0];
#pragma unroll
    for (int t = 0; t < 8; ++t) {
      v16bf b = *(const v16bf*)(wb + (t * 16 + nn) * 32 + half * 16);
      acc[t] = __builtin_amdgcn_wmma_f32_16x16x32_bf16(false, a, false, b,
                                                       (short)0, acc[t], false, false);
    }
    a = anext;
  }

  // stage strip in LDS, then 256B-contiguous ushort4 row stores
#pragma unroll
  for (int t = 0; t < 8; ++t)
#pragma unroll
    for (int r = 0; r < 8; ++r)
      outb[wave][r + half * 8][t * 16 + nn] = f2bf(acc[t][r]);
#pragma unroll
  for (int r = 0; r < 16; ++r) {
    ushort4 v = *(const ushort4*)&outb[wave][r][lane * 4];
    *(ushort4*)(out + (m0 + r) * D2_ + n0w + lane * 4) = v;
  }
}

// ---------------------------------------------------------------------------
// Kernel 4: zero-fill the w output region (only the 64-wide band is nonzero).
// ---------------------------------------------------------------------------
__global__ void zero_f4_kernel(float4* __restrict__ p, long long n4) {
  long long i = (long long)blockIdx.x * blockDim.x + threadIdx.x;
  const long long stride = (long long)gridDim.x * blockDim.x;
  const float4 z = make_float4(0.f, 0.f, 0.f, 0.f);
  for (; i < n4; i += stride) p[i] = z;
}

// ---------------------------------------------------------------------------
// Kernel 5: banded attention. One wave per 16-row tile (b, i0):
//   scores: 5 col tiles (j in [i0-64, i0+16)) x 16 K-chunks = 80 WMMAs
//   softmax over allowed band through LDS (stride-81 rows: conflict-free)
//   w band written coalesced + converted to bf16 A-fragments in LDS
//   retrieved = w @ x via 32 N-tiles x 3 K-chunks = 96 WMMAs (A from LDS)
// ---------------------------------------------------------------------------
__global__ __launch_bounds__(128) void band_attn_kernel(const unsigned short* __restrict__ qb,
                                                        const unsigned short* __restrict__ kb,
                                                        const unsigned short* __restrict__ xtb,
                                                        float* __restrict__ ret,
                                                        float* __restrict__ wout) {
  __shared__ float sc[4][16][81];
  __shared__ __align__(32) unsigned short wa[4][16][96];

  const int wave = threadIdx.x >> 5;
  const int lane = threadIdx.x & 31;
  const int half = lane >> 4;
  const int nn   = lane & 15;
  const int tIdx = blockIdx.x * 4 + wave;           // [0, 1024)
  const int b    = tIdx >> 8;                       // 256 tiles per batch
  const int i0   = (tIdx & 255) << 4;
  const size_t m0 = (size_t)tIdx * 16;              // == b*S_ + i0
  const int jbase = i0 - BAND;

  const v16bf zf = {};

  // ---- scores: Q(16x512) x K^T band ----
  v8f acc[5];
  {
    v8f z = {};
#pragma unroll
    for (int t = 0; t < 5; ++t) acc[t] = z;
  }
  for (int kc = 0; kc < 16; ++kc) {
    v16bf a = *(const v16bf*)(qb + (m0 + nn) * D2_ + kc * 32 + half * 16);
#pragma unroll
    for (int t = 0; t < 5; ++t) {
      const int j0 = jbase + t * 16;
      v16bf bf = zf;
      if (j0 >= 0)
        bf = *(const v16bf*)(kb + ((size_t)b * S_ + j0 + nn) * D2_ + kc * 32 + half * 16);
      acc[t] = __builtin_amdgcn_wmma_f32_16x16x32_bf16(false, a, false, bf,
                                                       (short)0, acc[t], false, false);
    }
  }
#pragma unroll
  for (int t = 0; t < 5; ++t)
#pragma unroll
    for (int r = 0; r < 8; ++r)
      sc[wave][r + half * 8][t * 16 + nn] = acc[t][r];

  // ---- banded softmax: lanes 0..15 own one row each ----
  if (lane < 16) {
    const int r = lane;
#pragma unroll
    for (int c = 0; c < 96; ++c) wa[wave][r][c] = 0;
    const int jlo = BAND - i0;                      // j >= 0 constraint
    const int clo = (r > jlo) ? r : jlo;            // j >= i-64 & j >= 0
    const int chi = r + BAND - 1;                   // j <= i-1
    if (clo <= chi) {
      float mx = -3.0e38f;
      for (int c = clo; c <= chi; ++c) mx = fmaxf(mx, sc[wave][r][c] * SCALE);
      float sum = 0.f;
      for (int c = clo; c <= chi; ++c) sum += __expf(sc[wave][r][c] * SCALE - mx);
      const float inv = 1.0f / sum;
      for (int c = clo; c <= chi; ++c) {
        float w = __expf(sc[wave][r][c] * SCALE - mx) * inv;
        sc[wave][r][c] = w;
        wa[wave][r][c] = f2bf(w);
      }
    }
  }
  // single wave owns its LDS slice; per-wave LDS ordering keeps this safe

  // ---- write the nonzero w band, 32 lanes sweep columns (coalesced) ----
  for (int r = 0; r < 16; ++r) {
    const int jlo = BAND - i0;
    const int clo = (r > jlo) ? r : jlo;
    const int chi = r + BAND - 1;
#pragma unroll
    for (int g = 0; g < 3; ++g) {
      int c = g * 32 + lane;
      if (c >= clo && c <= chi) {
        int j = jbase + c;
        wout[((size_t)b * S_ + i0 + r) * S_ + j] = sc[wave][r][c];
      }
    }
  }

  // ---- retrieved = w @ x : A-frag (bf16 w) from LDS, B-frag from x^T ----
  for (int nt = 0; nt < 32; ++nt) {
    v8f racc = {};
#pragma unroll
    for (int kc = 0; kc < 3; ++kc) {
      const int koff = kc * 32 + half * 16;
      v16bf a = *(const v16bf*)(&wa[wave][nn][koff]);
      v16bf bf = zf;
      const int jk = jbase + koff;
      if (jk >= 0 && koff < 80)                      // koff>=80 is zero-pad region
        bf = *(const v16bf*)(xtb + (size_t)(nt * 16 + nn) * MTOT + (size_t)b * S_ + jk);
      racc = __builtin_amdgcn_wmma_f32_16x16x32_bf16(false, a, false, bf,
                                                     (short)0, racc, false, false);
    }
    // 16-lane halves store 64B-contiguous f32 segments per row
#pragma unroll
    for (int r = 0; r < 8; ++r)
      ret[(m0 + r + half * 8) * D2_ + nt * 16 + nn] = racc[r];
  }
}

// ---------------------------------------------------------------------------
extern "C" void kernel_launch(void* const* d_in, const int* in_sizes, int n_in,
                              void* d_out, int out_size, void* d_ws, size_t ws_size,
                              hipStream_t stream) {
  const float* states = (const float*)d_in[0];
  const float* Wq     = (const float*)d_in[1];
  const float* Wk     = (const float*)d_in[2];

  char* ws = (char*)d_ws;
  // workspace layout (bytes), all offsets 256-aligned
  unsigned short* xb  = (unsigned short*)(ws + 0);          // 16 MB  x bf16 [MTOT,512]
  unsigned short* xtb = (unsigned short*)(ws + 16777216);   // 16 MB  x^T bf16 [512,MTOT]
  unsigned short* wqb = (unsigned short*)(ws + 33554432);   // 512 KB
  unsigned short* wkb = (unsigned short*)(ws + 34078720);   // 512 KB
  unsigned short* qb  = (unsigned short*)(ws + 34603008);   // 16 MB  q bf16
  unsigned short* kb  = (unsigned short*)(ws + 51380224);   // 16 MB  k bf16

  float* ret  = (float*)d_out;                              // [B,S,D,2] = MTOT*512 f32
  float* wout = ret + (size_t)MTOT * D2_;                   // [B,S,S] f32

  cvt_w_kernel<<<512, 256, 0, stream>>>(Wq, Wk, wqb, wkb);
  cvt_x_tr_kernel<<<dim3(MTOT / 32, D2_ / 32), dim3(32, 8), 0, stream>>>(states, xb, xtb);
  gemm_qk_kernel<<<dim3(MTOT / 16, 2), 128, 0, stream>>>(xb, wqb, wkb, qb, kb);
  zero_f4_kernel<<<2048, 256, 0, stream>>>((float4*)wout, (long long)MTOT * S_ / 4);
  band_attn_kernel<<<256, 128, 0, stream>>>(qb, kb, xtb, ret, wout);
}
